// BitMGQA_45174466020060
// MI455X (gfx1250) — compile-verified, matrix operand order checked by reading, not run on hardware
//
#include <hip/hip_runtime.h>
#include <hip/hip_bf16.h>


typedef __attribute__((ext_vector_type(8)))  int      v8i;
typedef __attribute__((ext_vector_type(8)))  float    v8f;
typedef __attribute__((ext_vector_type(16))) _Float16 v16h;

#define N_MODEL 1024
#define KV_MODEL 256
#define HEAD_DIM 64
#define KV_HEADS 4
#define GROUPS 4
#define BATCH 4
#define SEQ 2048
#define TOKENS (BATCH * SEQ)
#define ESTRIDE (SEQ + 4)

__device__ __forceinline__ int kb8(int v, int hl) {
  return ((v >> 1) << 4) | ((v & 1) << 2) | (hl << 3);
}
__device__ __forceinline__ int kb16(int v, int hl) {
  return ((v >> 2) << 4) | ((v & 3) << 1) | (hl << 3);
}

__global__ void wscale_kernel(const float* w0, int n0, const float* w1, int n1,
                              const float* w2, int n2, const float* w3, int n3,
                              float* wdq, float* wmul) {
  const float* w; int n;
  switch (blockIdx.x) {
    case 0:  w = w0; n = n0; break;
    case 1:  w = w1; n = n1; break;
    case 2:  w = w2; n = n2; break;
    default: w = w3; n = n3; break;
  }
  __shared__ float red[256];
  float s = 0.f;
  for (int i = threadIdx.x; i < n; i += 256) s += fabsf(w[i]);
  red[threadIdx.x] = s;
  __syncthreads();
  for (int st = 128; st > 0; st >>= 1) {
    if ((int)threadIdx.x < st) red[threadIdx.x] += red[threadIdx.x + st];
    __syncthreads();
  }
  if (threadIdx.x == 0) {
    float d = fmaxf(red[0] / (float)n, 1e-5f);
    wdq[blockIdx.x]  = d;
    wmul[blockIdx.x] = 1.0f / d;
  }
}

__global__ void wquant_kernel(const float* __restrict__ w, signed char* __restrict__ w8,
                              const float* __restrict__ wmul, int wIdx, int n) {
  float s = wmul[wIdx];
  for (int i = blockIdx.x * blockDim.x + threadIdx.x; i < n; i += gridDim.x * blockDim.x) {
    float q = fmaxf(-1.f, fminf(1.f, rintf(w[i] * s)));
    w8[i] = (signed char)(int)q;
  }
}

__global__ void foldq_kernel(const signed char* __restrict__ w8q, signed char* __restrict__ wsum8,
                             const float* __restrict__ bq, float* __restrict__ bqsum) {
  int r = blockIdx.x;
  int kv = r >> 6, d = r & 63;
  for (int i = threadIdx.x; i < N_MODEL; i += blockDim.x) {
    int s = 0;
#pragma unroll
    for (int g = 0; g < GROUPS; ++g)
      s += (int)w8q[(size_t)((kv * GROUPS + g) * HEAD_DIM + d) * N_MODEL + i];
    wsum8[(size_t)r * N_MODEL + i] = (signed char)s;
  }
  if (threadIdx.x == 0) {
    float s = 0.f;
    for (int g = 0; g < GROUPS; ++g) s += bq[(kv * GROUPS + g) * HEAD_DIM + d];
    bqsum[r] = s;
  }
}

__global__ void actq_kernel(const float* __restrict__ x, signed char* __restrict__ q8,
                            float* __restrict__ dx, int W) {
  int t = blockIdx.x;
  const float* row = x + (size_t)t * W;
  __shared__ float red[256];

  float ss = 0.f;
  for (int i = threadIdx.x; i < W; i += 256) { float v = row[i]; ss += v * v; }
  red[threadIdx.x] = ss;
  __syncthreads();
  for (int st = 128; st > 0; st >>= 1) {
    if ((int)threadIdx.x < st) red[threadIdx.x] += red[threadIdx.x + st];
    __syncthreads();
  }
  float rms = rsqrtf(red[0] / (float)W + 1e-6f);
  __syncthreads();

  float amax = 0.f;
  for (int i = threadIdx.x; i < W; i += 256) amax = fmaxf(amax, fabsf(row[i] * rms));
  red[threadIdx.x] = amax;
  __syncthreads();
  for (int st = 128; st > 0; st >>= 1) {
    if ((int)threadIdx.x < st) red[threadIdx.x] = fmaxf(red[threadIdx.x], red[threadIdx.x + st]);
    __syncthreads();
  }
  float scale = 127.0f / fmaxf(red[0], 1e-5f);

  for (int i = threadIdx.x; i < W; i += 256) {
    float q = rintf(row[i] * rms * scale);
    q = fmaxf(-128.f, fminf(127.f, q));
    q8[(size_t)t * W + i] = (signed char)(int)q;
  }
  if (threadIdx.x == 0) dx[t] = 1.0f / scale;
}

__global__ void gemm_iu8_kernel(const signed char* __restrict__ A8,
                                const float* __restrict__ dxA,
                                const signed char* __restrict__ B8,
                                const float* __restrict__ wdq, int wIdx,
                                const float* __restrict__ bias,
                                float* __restrict__ out,
                                int Kdim, int Nout) {
  int wave = threadIdx.x >> 5;
  int lane = threadIdx.x & 31;
  int hl = lane >> 4, l16 = lane & 15;
  int m0 = blockIdx.x * 16;
  int n0 = (blockIdx.y * 8 + wave) * 16;
  if (n0 >= Nout) return;

  const signed char* arow = A8 + (size_t)(m0 + l16) * Kdim;
  const signed char* brow = B8 + (size_t)(n0 + l16) * Kdim;

  v8i acc = {};
  for (int k0 = 0; k0 < Kdim; k0 += 64) {
    v8i a, b;
#pragma unroll
    for (int v = 0; v < 8; ++v) {
      int kb = k0 + kb8(v, hl);
      a[v] = *(const int*)(arow + kb);
      b[v] = *(const int*)(brow + kb);
    }
    acc = __builtin_amdgcn_wmma_i32_16x16x64_iu8(true, a, true, b, acc, false, false);
  }

  float dw = wdq[wIdx];
#pragma unroll
  for (int r = 0; r < 8; ++r) {
    int m = m0 + r + hl * 8;
    int n = n0 + l16;
    out[(size_t)m * Nout + n] = (float)acc[r] * dxA[m] * dw + bias[n];
  }
}

__global__ void attn_kernel(const float* __restrict__ Qs, const float* __restrict__ Kf,
                            const float* __restrict__ Vf, float* __restrict__ att,
                            float* __restrict__ ctx) {
  extern __shared__ float smem[];
  float* esh = smem;
  float* red = smem + 16 * ESTRIDE;

  int nt = blockIdx.x;
  int bh = blockIdx.y;
  int bb = bh >> 2, kv = bh & 3;
  int wave = threadIdx.x >> 5, lane = threadIdx.x & 31;
  int hl = lane >> 4, l16 = lane & 15;
  int n0 = nt * 16;
  size_t tokBase = (size_t)bb * SEQ;
  int dOff = kv * HEAD_DIM;

  v16h a0, a1;
  {
    const float* qrow = Qs + (tokBase + n0 + l16) * KV_MODEL + dOff;
#pragma unroll
    for (int v = 0; v < 8; ++v) {
      int k = kb16(v, hl);
      a0[2 * v]     = (_Float16)(qrow[k] * 0.125f);
      a0[2 * v + 1] = (_Float16)(qrow[k + 1] * 0.125f);
      a1[2 * v]     = (_Float16)(qrow[32 + k] * 0.125f);
      a1[2 * v + 1] = (_Float16)(qrow[32 + k + 1] * 0.125f);
    }
  }

  for (int st = wave; st < SEQ / 16; st += 4) {
    int s0 = st * 16;
    const float* krow = Kf + (tokBase + s0 + l16) * KV_MODEL + dOff;
    __builtin_prefetch(krow + 64 * KV_MODEL, 0, 1);
    v16h fb0, fb1;
#pragma unroll
    for (int v = 0; v < 8; ++v) {
      int k = kb16(v, hl);
      fb0[2 * v]     = (_Float16)krow[k];
      fb0[2 * v + 1] = (_Float16)krow[k + 1];
      fb1[2 * v]     = (_Float16)krow[32 + k];
      fb1[2 * v + 1] = (_Float16)krow[32 + k + 1];
    }
    v8f acc = {};
    acc = __builtin_amdgcn_wmma_f32_16x16x32_f16(false, a0, false, fb0, (short)0, acc, false, false);
    acc = __builtin_amdgcn_wmma_f32_16x16x32_f16(false, a1, false, fb1, (short)0, acc, false, false);
#pragma unroll
    for (int r = 0; r < 8; ++r)
      esh[(r + hl * 8) * ESTRIDE + s0 + l16] = acc[r];
  }
  __syncthreads();

  int row = threadIdx.x >> 3;
  int sub = threadIdx.x & 7;
  float* erow = esh + row * ESTRIDE;

  float m = -INFINITY;
  for (int i = sub * 256; i < sub * 256 + 256; ++i) m = fmaxf(m, erow[i]);
  red[threadIdx.x] = m;
  __syncthreads();
  if (sub == 0) {
    float mm = red[threadIdx.x];
    for (int j = 1; j < 8; ++j) mm = fmaxf(mm, red[threadIdx.x + j]);
    red[threadIdx.x] = mm;
  }
  __syncthreads();
  float rowmax = red[row * 8];

  float s = 0.f;
  for (int i = sub * 256; i < sub * 256 + 256; ++i) {
    float e = __expf(erow[i] - rowmax);
    erow[i] = e;
    s += e;
  }
  __syncthreads();
  red[threadIdx.x] = s;
  __syncthreads();
  if (sub == 0) {
    float t = 0.f;
    for (int j = 0; j < 8; ++j) t += red[threadIdx.x + j];
    red[threadIdx.x] = t;
  }
  __syncthreads();
  float inv = 1.0f / red[row * 8];
  for (int i = sub * 256; i < sub * 256 + 256; ++i) erow[i] *= inv;
  __syncthreads();

  float* attBase = att + ((size_t)bh * SEQ + n0) * SEQ;
  for (int idx = threadIdx.x; idx < 16 * SEQ; idx += 128) {
    int r = idx >> 11;
    int c = idx & (SEQ - 1);
    attBase[idx] = esh[r * ESTRIDE + c];
  }

  v8f cacc = {};
  int dsub = wave * 16;
  for (int k0 = 0; k0 < SEQ; k0 += 32) {
    __builtin_prefetch(Vf + (tokBase + k0 + 32 + (size_t)l16) * KV_MODEL + dOff + dsub, 0, 1);
    v16h pa, vb;
#pragma unroll
    for (int v = 0; v < 8; ++v) {
      int k = kb16(v, hl);
      pa[2 * v]     = (_Float16)esh[l16 * ESTRIDE + k0 + k];
      pa[2 * v + 1] = (_Float16)esh[l16 * ESTRIDE + k0 + k + 1];
      const float* vr = Vf + (tokBase + k0 + k) * KV_MODEL + dOff + dsub + l16;
      vb[2 * v]     = (_Float16)vr[0];
      vb[2 * v + 1] = (_Float16)vr[KV_MODEL];
    }
    cacc = __builtin_amdgcn_wmma_f32_16x16x32_f16(false, pa, false, vb, (short)0, cacc, false, false);
  }
#pragma unroll
  for (int r = 0; r < 8; ++r) {
    int mrow = n0 + r + hl * 8;
    ctx[(tokBase + mrow) * KV_MODEL + dOff + dsub + l16] = cacc[r];
  }
}

static inline size_t alignUp(size_t x) { return (x + 255) & ~(size_t)255; }

extern "C" void kernel_launch(void* const* d_in, const int* in_sizes, int n_in,
                              void* d_out, int out_size, void* d_ws, size_t ws_size,
                              hipStream_t stream) {
  (void)in_sizes; (void)n_in; (void)out_size; (void)ws_size;

  const float* query = (const float*)d_in[0];
  const float* key   = (const float*)d_in[1];
  const float* value = (const float*)d_in[2];
  const float* Wq    = (const float*)d_in[3];
  const float* bq    = (const float*)d_in[4];
  const float* Wk    = (const float*)d_in[5];
  const float* bk    = (const float*)d_in[6];
  const float* Wv    = (const float*)d_in[7];
  const float* bv    = (const float*)d_in[8];
  const float* Wo    = (const float*)d_in[9];
  const float* bo    = (const float*)d_in[10];

  float* xOut   = (float*)d_out;
  float* attOut = (float*)d_out + (size_t)TOKENS * N_MODEL;

  char* base = (char*)d_ws;
  size_t off = 0;
  auto alloc = [&](size_t bytes) -> void* {
    void* p = base + off;
    off = alignUp(off + bytes);
    return p;
  };
  signed char* q8q   = (signed char*)alloc((size_t)TOKENS * N_MODEL);
  signed char* q8k   = (signed char*)alloc((size_t)TOKENS * N_MODEL);
  signed char* q8v   = (signed char*)alloc((size_t)TOKENS * N_MODEL);
  float* dxq         = (float*)alloc((size_t)TOKENS * 4);
  float* dxk         = (float*)alloc((size_t)TOKENS * 4);
  float* dxv         = (float*)alloc((size_t)TOKENS * 4);
  signed char* w8q   = (signed char*)alloc((size_t)N_MODEL * N_MODEL);
  signed char* w8k   = (signed char*)alloc((size_t)KV_MODEL * N_MODEL);
  signed char* w8v   = (signed char*)alloc((size_t)KV_MODEL * N_MODEL);
  signed char* w8o   = (signed char*)alloc((size_t)N_MODEL * KV_MODEL);
  signed char* wsum8 = (signed char*)alloc((size_t)KV_MODEL * N_MODEL);
  float* wdq         = (float*)alloc(4 * 4);
  float* wmul        = (float*)alloc(4 * 4);
  float* bqsum       = (float*)alloc(KV_MODEL * 4);
  float* Qs          = (float*)alloc((size_t)TOKENS * KV_MODEL * 4);
  float* Kf          = (float*)alloc((size_t)TOKENS * KV_MODEL * 4);
  float* Vf          = (float*)alloc((size_t)TOKENS * KV_MODEL * 4);
  float* ctx         = (float*)alloc((size_t)TOKENS * KV_MODEL * 4);
  signed char* ctx8  = (signed char*)alloc((size_t)TOKENS * KV_MODEL);
  float* dxo         = (float*)alloc((size_t)TOKENS * 4);

  wscale_kernel<<<4, 256, 0, stream>>>(Wq, N_MODEL * N_MODEL, Wk, KV_MODEL * N_MODEL,
                                       Wv, KV_MODEL * N_MODEL, Wo, N_MODEL * KV_MODEL,
                                       wdq, wmul);
  wquant_kernel<<<512, 256, 0, stream>>>(Wq, w8q, wmul, 0, N_MODEL * N_MODEL);
  wquant_kernel<<<256, 256, 0, stream>>>(Wk, w8k, wmul, 1, KV_MODEL * N_MODEL);
  wquant_kernel<<<256, 256, 0, stream>>>(Wv, w8v, wmul, 2, KV_MODEL * N_MODEL);
  wquant_kernel<<<256, 256, 0, stream>>>(Wo, w8o, wmul, 3, N_MODEL * KV_MODEL);
  foldq_kernel<<<KV_MODEL, 256, 0, stream>>>(w8q, wsum8, bq, bqsum);
  actq_kernel<<<TOKENS, 256, 0, stream>>>(query, q8q, dxq, N_MODEL);
  actq_kernel<<<TOKENS, 256, 0, stream>>>(key,   q8k, dxk, N_MODEL);
  actq_kernel<<<TOKENS, 256, 0, stream>>>(value, q8v, dxv, N_MODEL);
  gemm_iu8_kernel<<<dim3(TOKENS / 16, KV_MODEL / 128), 256, 0, stream>>>(
      q8q, dxq, wsum8, wdq, 0, bqsum, Qs, N_MODEL, KV_MODEL);
  gemm_iu8_kernel<<<dim3(TOKENS / 16, KV_MODEL / 128), 256, 0, stream>>>(
      q8k, dxk, w8k, wdq, 1, bk, Kf, N_MODEL, KV_MODEL);
  gemm_iu8_kernel<<<dim3(TOKENS / 16, KV_MODEL / 128), 256, 0, stream>>>(
      q8v, dxv, w8v, wdq, 2, bv, Vf, N_MODEL, KV_MODEL);
  const int ATT_SMEM = (16 * ESTRIDE + 128) * (int)sizeof(float);
  (void)hipFuncSetAttribute((const void*)attn_kernel,
                            hipFuncAttributeMaxDynamicSharedMemorySize, ATT_SMEM);
  attn_kernel<<<dim3(SEQ / 16, BATCH * KV_HEADS), 128, ATT_SMEM, stream>>>(
      Qs, Kf, Vf, attOut, ctx);
  actq_kernel<<<TOKENS, 256, 0, stream>>>(ctx, ctx8, dxo, KV_MODEL);
  gemm_iu8_kernel<<<dim3(TOKENS / 16, N_MODEL / 128), 256, 0, stream>>>(
      ctx8, dxo, w8o, wdq, 3, bo, xOut, KV_MODEL, N_MODEL);
}
